// MusicVAE_89335319757362
// MI455X (gfx1250) — matrix operand order repaired and checked by reading, not compiled, not run on hardware
//
#include <hip/hip_runtime.h>
#include <cstdint>
#include <cstddef>

// ---------------- problem constants ----------------
#define T_SEQ 64
#define BATCH 32
#define DIN   512
#define ENCH  2048
#define LATD  512
#define CONH  1024
#define DECH  1024

typedef __bf16 bf16_t;
typedef __attribute__((ext_vector_type(16))) __bf16 v16bf;
typedef __attribute__((ext_vector_type(8)))  __bf16 v8bf;
typedef __attribute__((ext_vector_type(8)))  float  v8f;

// ---------------- device helpers ----------------
static __device__ __forceinline__ bf16_t f2bf(float f) {
  union { float f; uint32_t u; } v; v.f = f;
  uint32_t r = v.u + 0x7fffu + ((v.u >> 16) & 1u);   // round-to-nearest-even
  uint16_t h = (uint16_t)(r >> 16);
  return __builtin_bit_cast(bf16_t, h);
}

static __device__ __forceinline__ v8bf load8bf(const bf16_t* p) {
  return *(const v8bf*)p;   // 16B aligned by construction
}

// A fragment (16x32 bf16, MxK). Lane L(0-15): row M=L, K={k0..k0+7, k0+16..k0+23};
// Lane L+16: row M=L, K={k0+8..k0+15, k0+24..k0+31}. (ISA 7.12.2)
static __device__ __forceinline__ v16bf make_afrag(const bf16_t* row, int k0, int lane) {
  int kb = k0 + ((lane >> 4) << 3);
  v8bf lo = load8bf(row + kb);
  v8bf hi = load8bf(row + kb + 16);
  v16bf a;
#pragma unroll
  for (int i = 0; i < 8; ++i) { a[i] = lo[i]; a[i + 8] = hi[i]; }
  return a;
}

// B fragment (32x16 bf16, KxN), B = W^T with W row-major [N,K].
// Lanes 0-15 hold K={k0..k0+7, k0+16..k0+23} of column N=lane; lanes 16-31 hold K={k0+8..k0+15, k0+24..k0+31}.
static __device__ __forceinline__ v16bf make_bfrag(const bf16_t* wrow, int k0, int lane) {
  const bf16_t* p = wrow + k0 + ((lane >> 4) << 3);
  v8bf lo = load8bf(p);
  v8bf hi = load8bf(p + 16);
  v16bf b;
#pragma unroll
  for (int i = 0; i < 8; ++i) { b[i] = lo[i]; b[i + 8] = hi[i]; }
  return b;
}

static __device__ __forceinline__ v8f wmma_bf16(v16bf a, v16bf b, v8f c) {
  return __builtin_amdgcn_wmma_f32_16x16x32_bf16(false, a, false, b, (short)0, c,
                                                 false, false);
}

static __device__ __forceinline__ float sigmoidf_(float x) {
  return 1.0f / (1.0f + __expf(-x));
}
static __device__ __forceinline__ float softplusf_(float x) {
  if (x > 20.0f)  return x;
  if (x < -20.0f) return __expf(x);
  return log1pf(__expf(x));
}

// ---------------- generic bf16 GEMM: C[M,N] = A[M,K] @ W[N,K]^T (+bias) -------
// wave tile 32(M) x 64(N); block = 8 waves along N (512 cols); grid.x=N/512, grid.y=M/32
__global__ void gemm_bf16(const bf16_t* __restrict__ A, int lda,
                          const bf16_t* __restrict__ W, int ldw,
                          const float* __restrict__ bias,
                          float* __restrict__ C, int ldc, int K) {
  int lane = threadIdx.x & 31;
  int wave = threadIdx.x >> 5;
  int m0 = blockIdx.y * 32;
  int n0 = blockIdx.x * 512 + wave * 64;

  v8f zero = {0.f, 0.f, 0.f, 0.f, 0.f, 0.f, 0.f, 0.f};
  v8f acc[2][4];
#pragma unroll
  for (int mt = 0; mt < 2; ++mt)
#pragma unroll
    for (int nt = 0; nt < 4; ++nt) acc[mt][nt] = zero;

  const bf16_t* arow0 = A + (size_t)(m0 + (lane & 15)) * lda;
  const bf16_t* arow1 = arow0 + (size_t)16 * lda;

  for (int k0 = 0; k0 < K; k0 += 32) {
    v16bf a0 = make_afrag(arow0, k0, lane);
    v16bf a1 = make_afrag(arow1, k0, lane);
#pragma unroll
    for (int nt = 0; nt < 4; ++nt) {
      const bf16_t* wrow = W + (size_t)(n0 + nt * 16 + (lane & 15)) * ldw;
      __builtin_prefetch(wrow + k0 + 64, 0, 1);   // global_prefetch_b8
      v16bf b = make_bfrag(wrow, k0, lane);
      acc[0][nt] = wmma_bf16(a0, b, acc[0][nt]);
      acc[1][nt] = wmma_bf16(a1, b, acc[1][nt]);
    }
  }
  int mr = (lane >> 4) * 8, nc = lane & 15;
#pragma unroll
  for (int mt = 0; mt < 2; ++mt)
#pragma unroll
    for (int nt = 0; nt < 4; ++nt) {
      int n = n0 + nt * 16 + nc;
      float bb = bias ? bias[n] : 0.f;
#pragma unroll
      for (int v = 0; v < 8; ++v)
        C[(size_t)(m0 + mt * 16 + mr + v) * ldc + n] = acc[mt][nt][v] + bb;
    }
}

// ---------------- fused LSTM cell step -----------------------------------------
// pre = ginA + ginB + bias + X@Wih^T + Hprev@Whh^T  (gate order i,f,g,o)
// c' = sig(f)*c + sig(i)*tanh(g);  h' = sig(o)*tanh(c')
struct DirArgs {
  const float*  ginA;      // [B,4H] or null (precomputed input projection, no bias)
  const float*  ginB;      // [B,4H] or null
  const bf16_t* x;         // [B,Kin] row stride Kin, or null
  const bf16_t* wih;       // [4H,Kin]
  const bf16_t* hprev;     // [B,*] row stride hp_stride
  int           hp_stride;
  const bf16_t* whh;       // [4H,H]
  const float*  bias;      // [4H] = bih+bhh
  const float*  cprev;     // [B,H]
  float*        cnew;      // [B,H] (may alias cprev's buffer across steps)
  bf16_t*       hnew;      // row stride hn_stride
  int           hn_stride;
};

// grid: x = H/128 (8 waves * 16 hidden each), y = ndir; block = 256
__global__ void lstm_step(DirArgs d0, DirArgs d1, int H, int Kin) {
  const DirArgs a = (blockIdx.y == 0) ? d0 : d1;
  int lane = threadIdx.x & 31;
  int wave = threadIdx.x >> 5;
  int n0 = (blockIdx.x * 8 + wave) * 16;

  v8f zero = {0.f, 0.f, 0.f, 0.f, 0.f, 0.f, 0.f, 0.f};
  v8f acc[4][2];
#pragma unroll
  for (int g = 0; g < 4; ++g) { acc[g][0] = zero; acc[g][1] = zero; }

  // recurrent part: Hprev[B,H] @ Whh[4H,H]^T
  {
    const bf16_t* h0row = a.hprev + (size_t)(lane & 15) * a.hp_stride;
    const bf16_t* h1row = a.hprev + (size_t)(16 + (lane & 15)) * a.hp_stride;
    for (int k0 = 0; k0 < H; k0 += 32) {
      v16bf af0 = make_afrag(h0row, k0, lane);
      v16bf af1 = make_afrag(h1row, k0, lane);
#pragma unroll
      for (int g = 0; g < 4; ++g) {
        const bf16_t* wrow = a.whh + (size_t)(g * H + n0 + (lane & 15)) * H;
        __builtin_prefetch(wrow + k0 + 64, 0, 1);
        v16bf b = make_bfrag(wrow, k0, lane);
        acc[g][0] = wmma_bf16(af0, b, acc[g][0]);
        acc[g][1] = wmma_bf16(af1, b, acc[g][1]);
      }
    }
  }
  // optional dynamic input part: X[B,Kin] @ Wih[4H,Kin]^T (uniform branch)
  if (a.x) {
    const bf16_t* x0 = a.x + (size_t)(lane & 15) * Kin;
    const bf16_t* x1 = a.x + (size_t)(16 + (lane & 15)) * Kin;
    for (int k0 = 0; k0 < Kin; k0 += 32) {
      v16bf af0 = make_afrag(x0, k0, lane);
      v16bf af1 = make_afrag(x1, k0, lane);
#pragma unroll
      for (int g = 0; g < 4; ++g) {
        const bf16_t* wrow = a.wih + (size_t)(g * H + n0 + (lane & 15)) * Kin;
        v16bf b = make_bfrag(wrow, k0, lane);
        acc[g][0] = wmma_bf16(af0, b, acc[g][0]);
        acc[g][1] = wmma_bf16(af1, b, acc[g][1]);
      }
    }
  }

  int mr = (lane >> 4) * 8, nc = lane & 15;
  int n = n0 + nc;
#pragma unroll
  for (int mt = 0; mt < 2; ++mt)
#pragma unroll
    for (int v = 0; v < 8; ++v) {
      int b = mt * 16 + mr + v;                 // batch row
      size_t g4 = (size_t)b * 4 * H;
      float pi = acc[0][mt][v] + a.bias[0 * H + n];
      float pf = acc[1][mt][v] + a.bias[1 * H + n];
      float pg = acc[2][mt][v] + a.bias[2 * H + n];
      float po = acc[3][mt][v] + a.bias[3 * H + n];
      if (a.ginA) {
        pi += a.ginA[g4 + 0 * H + n]; pf += a.ginA[g4 + 1 * H + n];
        pg += a.ginA[g4 + 2 * H + n]; po += a.ginA[g4 + 3 * H + n];
      }
      if (a.ginB) {
        pi += a.ginB[g4 + 0 * H + n]; pf += a.ginB[g4 + 1 * H + n];
        pg += a.ginB[g4 + 2 * H + n]; po += a.ginB[g4 + 3 * H + n];
      }
      float ii = sigmoidf_(pi), ff = sigmoidf_(pf);
      float gg = tanhf(pg),     oo = sigmoidf_(po);
      float c = ff * a.cprev[(size_t)b * H + n] + ii * gg;
      a.cnew[(size_t)b * H + n] = c;
      a.hnew[(size_t)b * a.hn_stride + n] = f2bf(oo * tanhf(c));
    }
}

// ---------------- small elementwise kernels ------------------------------------
__global__ void k_f32_to_bf16(const float* __restrict__ s, bf16_t* __restrict__ d, int n) {
  int i = blockIdx.x * blockDim.x + threadIdx.x;
  if (i < n) d[i] = f2bf(s[i]);
}
__global__ void k_bias_sum(const float* __restrict__ a, const float* __restrict__ b,
                           float* __restrict__ d, int n) {
  int i = blockIdx.x * blockDim.x + threadIdx.x;
  if (i < n) d[i] = a[i] + b[i];
}
__global__ void k_zero_bf16(bf16_t* p, int n) {
  int i = blockIdx.x * blockDim.x + threadIdx.x;
  if (i < n) p[i] = f2bf(0.f);
}
__global__ void k_zero_f32(float* p, int n) {
  int i = blockIdx.x * blockDim.x + threadIdx.x;
  if (i < n) p[i] = 0.f;
}
// xs[t,b,d] = x[b,t,d] ; xprev[t,b,d] = (t==0) ? 0 : x[b,t-1,d]
__global__ void k_xs_prep(const float* __restrict__ x, bf16_t* __restrict__ xs,
                          bf16_t* __restrict__ xprev, int n) {
  int i = blockIdx.x * blockDim.x + threadIdx.x;
  if (i >= n) return;
  int d = i & (DIN - 1);
  int b = (i >> 9) & (BATCH - 1);
  int t = i >> 14;
  xs[i]    = f2bf(x[((size_t)b * T_SEQ + t) * DIN + d]);
  xprev[i] = (t == 0) ? f2bf(0.f) : f2bf(x[((size_t)b * T_SEQ + (t - 1)) * DIN + d]);
}
// hcat[b, 0:2048]=Y0[63,b,0:2048]; [2048:4096]=Y0[0,b,2048:4096]; same for Y1
__global__ void k_hcat(const bf16_t* __restrict__ y0, const bf16_t* __restrict__ y1,
                       bf16_t* __restrict__ hcat, int n) {
  int i = blockIdx.x * blockDim.x + threadIdx.x;
  if (i >= n) return;
  int j = i & 8191;
  int b = i >> 13;
  bf16_t v;
  if (j < 4096) {
    const bf16_t* base = (j < 2048) ? (y0 + (size_t)(T_SEQ - 1) * BATCH * 4096) : y0;
    v = base[(size_t)b * 4096 + j];
  } else {
    int jj = j - 4096;
    const bf16_t* base = (jj < 2048) ? (y1 + (size_t)(T_SEQ - 1) * BATCH * 4096) : y1;
    v = base[(size_t)b * 4096 + jj];
  }
  hcat[i] = v;
}
// LayerNorm(mu_raw) -> mu ; softplus(std_raw) -> std ; z = mu + std*eps
__global__ void k_latent(const float* __restrict__ mu_raw, const float* __restrict__ std_raw,
                         const float* __restrict__ eps, float* __restrict__ mu_out,
                         float* __restrict__ std_out, float* __restrict__ z_f,
                         bf16_t* __restrict__ z_bf) {
  int b = blockIdx.x, tid = threadIdx.x;
  const float* M = mu_raw + (size_t)b * LATD;
  __shared__ float red[256];
  float v0 = M[tid], v1 = M[tid + 256];
  red[tid] = v0 + v1;
  __syncthreads();
  for (int s = 128; s > 0; s >>= 1) { if (tid < s) red[tid] += red[tid + s]; __syncthreads(); }
  float mean = red[0] / (float)LATD;
  __syncthreads();
  float d0 = v0 - mean, d1 = v1 - mean;
  red[tid] = d0 * d0 + d1 * d1;
  __syncthreads();
  for (int s = 128; s > 0; s >>= 1) { if (tid < s) red[tid] += red[tid + s]; __syncthreads(); }
  float rs = rsqrtf(red[0] / (float)LATD + 1e-5f);
  for (int j = tid; j < LATD; j += 256) {
    size_t idx = (size_t)b * LATD + j;
    float mn = (M[j] - mean) * rs;
    float st = softplusf_(std_raw[idx]);
    float z  = mn + st * eps[idx];
    mu_out[idx] = mn; std_out[idx] = st; z_f[idx] = z; z_bf[idx] = f2bf(z);
  }
}
// hz[b,n] = src[b, n%512]  (concat(z,z)); write bf16 + f32
__global__ void k_make_hz(const float* __restrict__ src, bf16_t* __restrict__ hbf,
                          float* __restrict__ hf, int n) {
  int i = blockIdx.x * blockDim.x + threadIdx.x;
  if (i >= n) return;
  int nn = i & 1023;
  int b = i >> 10;
  float v = src[(size_t)b * LATD + (nn & 511)];
  hbf[i] = f2bf(v); hf[i] = v;
}
// softmax over 512 logits per (t,b); write transposed into out[b,t,:]
__global__ void k_softmax_out(const float* __restrict__ logits, float* __restrict__ out) {
  int row = blockIdx.x;
  int t = row / BATCH, b = row % BATCH;
  const float* L = logits + (size_t)row * DIN;
  float* O = out + ((size_t)b * T_SEQ + t) * DIN;
  __shared__ float red[256];
  int tid = threadIdx.x;
  float v0 = L[tid], v1 = L[tid + 256];
  red[tid] = fmaxf(v0, v1);
  __syncthreads();
  for (int s = 128; s > 0; s >>= 1) { if (tid < s) red[tid] = fmaxf(red[tid], red[tid + s]); __syncthreads(); }
  float mx = red[0];
  __syncthreads();
  float e0 = __expf(v0 - mx), e1 = __expf(v1 - mx);
  red[tid] = e0 + e1;
  __syncthreads();
  for (int s = 128; s > 0; s >>= 1) { if (tid < s) red[tid] += red[tid + s]; __syncthreads(); }
  float inv = 1.0f / red[0];
  O[tid] = e0 * inv; O[tid + 256] = e1 * inv;
}

// ---------------- host orchestration -------------------------------------------
extern "C" void kernel_launch(void* const* d_in, const int* in_sizes, int n_in,
                              void* d_out, int out_size, void* d_ws, size_t ws_size,
                              hipStream_t stream) {
  (void)in_sizes; (void)n_in; (void)out_size; (void)ws_size;
  const float* x_in  = (const float*)d_in[0];
  const float* eps   = (const float*)d_in[1];
  // enc l0_f:2-5  l0_b:6-9  l1_f:10-13  l1_b:14-17  mu:18-19  std:20-21
  // con l0:22-25  l1:26-29  lin:30-31
  // dec l0:32-35  l1:36-39  out:40-41
  auto F = [&](int i) { return (const float*)d_in[i]; };

  // ---- bump allocator on d_ws ----
  char* wsp = (char*)d_ws;
  size_t off = 0;
  auto alloc = [&](size_t bytes) -> void* {
    void* p = wsp + off;
    off = (off + bytes + 255) & ~(size_t)255;
    return p;
  };
  auto abf = [&](size_t elems) { return (bf16_t*)alloc(elems * sizeof(bf16_t)); };
  auto af32 = [&](size_t elems) { return (float*)alloc(elems * sizeof(float)); };

  // bf16 weights
  bf16_t *wE0fI = abf(8192u * 512),  *wE0fH = abf(8192u * 2048);
  bf16_t *wE0bI = abf(8192u * 512),  *wE0bH = abf(8192u * 2048);
  bf16_t *wE1fI = abf(8192u * 4096), *wE1fH = abf(8192u * 2048);
  bf16_t *wE1bI = abf(8192u * 4096), *wE1bH = abf(8192u * 2048);
  bf16_t *wMu = abf(512u * 8192), *wStd = abf(512u * 8192);
  bf16_t *wC0I = abf(4096u * 512),  *wC0H = abf(4096u * 1024);
  bf16_t *wC1I = abf(4096u * 1024), *wC1H = abf(4096u * 1024);
  bf16_t *wLin = abf(512u * 1024);
  bf16_t *wD0I = abf(4096u * 1024), *wD0H = abf(4096u * 1024);
  bf16_t *wD1I = abf(4096u * 1024), *wD1H = abf(4096u * 1024);
  bf16_t *wOut = abf(512u * 1024);
  // summed biases (bih+bhh), f32
  float *bE0f = af32(8192), *bE0b = af32(8192), *bE1f = af32(8192), *bE1b = af32(8192);
  float *bC0 = af32(4096), *bC1 = af32(4096), *bD0 = af32(4096), *bD1 = af32(4096);
  // activations
  bf16_t* xs_bf    = abf((size_t)T_SEQ * BATCH * DIN);
  bf16_t* xprev_bf = abf((size_t)T_SEQ * BATCH * DIN);
  float*  ginF = af32((size_t)T_SEQ * BATCH * 8192);   // reused: enc l0f -> l1f -> xproj
  float*  ginB = af32((size_t)T_SEQ * BATCH * 8192);   // reused: enc l0b -> l1b -> logits
  bf16_t* Y0 = abf((size_t)T_SEQ * BATCH * 4096);
  bf16_t* Y1 = abf((size_t)T_SEQ * BATCH * 4096);
  float *ce0f = af32(BATCH * ENCH), *ce0b = af32(BATCH * ENCH);
  float *ce1f = af32(BATCH * ENCH), *ce1b = af32(BATCH * ENCH);
  bf16_t* zeros_bf = abf((size_t)BATCH * 4096);
  float*  zeros_f  = af32((size_t)BATCH * ENCH);
  bf16_t* hcat_bf  = abf((size_t)BATCH * 8192);
  float *mu_raw = af32(BATCH * LATD), *std_raw = af32(BATCH * LATD), *z_f = af32(BATCH * LATD);
  bf16_t* z_bf = abf(BATCH * LATD);
  float* ginz = af32((size_t)BATCH * 4096);
  bf16_t *Yc0 = abf(4u * BATCH * CONH), *Yc1 = abf(4u * BATCH * CONH);
  float *cc0 = af32(BATCH * CONH), *cc1 = af32(BATCH * CONH);
  bf16_t* hz_bf = abf(BATCH * CONH);  float* hz_f = af32(BATCH * CONH);
  float*  feat = af32(4u * BATCH * LATD);
  bf16_t* feat_bf = abf(4u * BATCH * LATD);
  float*  zproj = af32(4u * BATCH * 4096);
  bf16_t *Yd0 = abf((size_t)T_SEQ * BATCH * DECH), *Yd1 = abf((size_t)T_SEQ * BATCH * DECH);
  float *cd0 = af32(BATCH * DECH), *cd1 = af32(BATCH * DECH);
  bf16_t* hzd_bf = abf(BATCH * DECH);  float* hzd_f = af32(BATCH * DECH);
  float* xproj  = ginF;   // alias (free after encoder)
  float* logits = ginB;   // alias (free after encoder)

  auto lgrid = [](int n) { return dim3((unsigned)((n + 255) / 256)); };
  auto conv = [&](const float* s, bf16_t* d, int n) {
    k_f32_to_bf16<<<lgrid(n), 256, 0, stream>>>(s, d, n);
  };
  auto bsum = [&](int ia, int ib, float* d, int n) {
    k_bias_sum<<<lgrid(n), 256, 0, stream>>>(F(ia), F(ib), d, n);
  };
  auto gemm = [&](const bf16_t* A, int lda, const bf16_t* W, int ldw, const float* bias,
                  float* C, int ldc, int M, int N, int K) {
    gemm_bf16<<<dim3((unsigned)(N / 512), (unsigned)(M / 32)), 256, 0, stream>>>(
        A, lda, W, ldw, bias, C, ldc, K);
  };

  // ---- setup: zeros, weight conversion, bias sums, input transpose ----
  k_zero_bf16<<<lgrid(BATCH * 4096), 256, 0, stream>>>(zeros_bf, BATCH * 4096);
  k_zero_f32<<<lgrid(BATCH * ENCH), 256, 0, stream>>>(zeros_f, BATCH * ENCH);
  conv(F(2), wE0fI, 8192 * 512);    conv(F(3), wE0fH, 8192 * 2048);
  conv(F(6), wE0bI, 8192 * 512);    conv(F(7), wE0bH, 8192 * 2048);
  conv(F(10), wE1fI, 8192 * 4096);  conv(F(11), wE1fH, 8192 * 2048);
  conv(F(14), wE1bI, 8192 * 4096);  conv(F(15), wE1bH, 8192 * 2048);
  conv(F(18), wMu, 512 * 8192);     conv(F(20), wStd, 512 * 8192);
  conv(F(22), wC0I, 4096 * 512);    conv(F(23), wC0H, 4096 * 1024);
  conv(F(26), wC1I, 4096 * 1024);   conv(F(27), wC1H, 4096 * 1024);
  conv(F(30), wLin, 512 * 1024);
  conv(F(32), wD0I, 4096 * 1024);   conv(F(33), wD0H, 4096 * 1024);
  conv(F(36), wD1I, 4096 * 1024);   conv(F(37), wD1H, 4096 * 1024);
  conv(F(40), wOut, 512 * 1024);
  bsum(4, 5, bE0f, 8192);  bsum(8, 9, bE0b, 8192);
  bsum(12, 13, bE1f, 8192); bsum(16, 17, bE1b, 8192);
  bsum(24, 25, bC0, 4096); bsum(28, 29, bC1, 4096);
  bsum(34, 35, bD0, 4096); bsum(38, 39, bD1, 4096);
  k_xs_prep<<<lgrid(T_SEQ * BATCH * DIN), 256, 0, stream>>>(x_in, xs_bf, xprev_bf,
                                                            T_SEQ * BATCH * DIN);

  // ---- encoder layer 0: precompute input projections, then recurrence ----
  gemm(xs_bf, DIN, wE0fI, DIN, nullptr, ginF, 8192, T_SEQ * BATCH, 8192, DIN);
  gemm(xs_bf, DIN, wE0bI, DIN, nullptr, ginB, 8192, T_SEQ * BATCH, 8192, DIN);
  for (int t = 0; t < T_SEQ; ++t) {
    int tb = T_SEQ - 1 - t;
    DirArgs f{}, bw{};
    f.ginA = ginF + (size_t)t * BATCH * 8192;
    f.hprev = (t == 0) ? zeros_bf : Y0 + (size_t)(t - 1) * BATCH * 4096;
    f.hp_stride = 4096;
    f.whh = wE0fH; f.bias = bE0f;
    f.cprev = (t == 0) ? zeros_f : ce0f; f.cnew = ce0f;
    f.hnew = Y0 + (size_t)t * BATCH * 4096; f.hn_stride = 4096;
    bw.ginA = ginB + (size_t)tb * BATCH * 8192;
    bw.hprev = (t == 0) ? zeros_bf : Y0 + (size_t)(tb + 1) * BATCH * 4096 + 2048;
    bw.hp_stride = 4096;
    bw.whh = wE0bH; bw.bias = bE0b;
    bw.cprev = (t == 0) ? zeros_f : ce0b; bw.cnew = ce0b;
    bw.hnew = Y0 + (size_t)tb * BATCH * 4096 + 2048; bw.hn_stride = 4096;
    lstm_step<<<dim3(ENCH / 128, 2), 256, 0, stream>>>(f, bw, ENCH, 0);
  }

  // ---- encoder layer 1 (input = concat(yf,yb) = Y0 rows) ----
  gemm(Y0, 4096, wE1fI, 4096, nullptr, ginF, 8192, T_SEQ * BATCH, 8192, 4096);
  gemm(Y0, 4096, wE1bI, 4096, nullptr, ginB, 8192, T_SEQ * BATCH, 8192, 4096);
  for (int t = 0; t < T_SEQ; ++t) {
    int tb = T_SEQ - 1 - t;
    DirArgs f{}, bw{};
    f.ginA = ginF + (size_t)t * BATCH * 8192;
    f.hprev = (t == 0) ? zeros_bf : Y1 + (size_t)(t - 1) * BATCH * 4096;
    f.hp_stride = 4096;
    f.whh = wE1fH; f.bias = bE1f;
    f.cprev = (t == 0) ? zeros_f : ce1f; f.cnew = ce1f;
    f.hnew = Y1 + (size_t)t * BATCH * 4096; f.hn_stride = 4096;
    bw.ginA = ginB + (size_t)tb * BATCH * 8192;
    bw.hprev = (t == 0) ? zeros_bf : Y1 + (size_t)(tb + 1) * BATCH * 4096 + 2048;
    bw.hp_stride = 4096;
    bw.whh = wE1bH; bw.bias = bE1b;
    bw.cprev = (t == 0) ? zeros_f : ce1b; bw.cnew = ce1b;
    bw.hnew = Y1 + (size_t)tb * BATCH * 4096 + 2048; bw.hn_stride = 4096;
    lstm_step<<<dim3(ENCH / 128, 2), 256, 0, stream>>>(f, bw, ENCH, 0);
  }

  // ---- latent head ----
  float* out_xprob = (float*)d_out;
  float* out_mu  = out_xprob + (size_t)BATCH * T_SEQ * DIN;
  float* out_std = out_mu + (size_t)BATCH * LATD;
  k_hcat<<<lgrid(BATCH * 8192), 256, 0, stream>>>(Y0, Y1, hcat_bf, BATCH * 8192);
  gemm(hcat_bf, 8192, wMu, 8192, F(19), mu_raw, LATD, BATCH, LATD, 8192);
  gemm(hcat_bf, 8192, wStd, 8192, F(21), std_raw, LATD, BATCH, LATD, 8192);
  k_latent<<<dim3(BATCH), 256, 0, stream>>>(mu_raw, std_raw, eps, out_mu, out_std, z_f, z_bf);

  // ---- conductor ----
  k_make_hz<<<lgrid(BATCH * CONH), 256, 0, stream>>>(z_f, hz_bf, hz_f, BATCH * CONH);
  gemm(z_bf, LATD, wC0I, LATD, nullptr, ginz, 4096, BATCH, 4096, LATD);
  for (int s = 0; s < 4; ++s) {
    DirArgs c0{};
    c0.ginA = ginz;
    c0.hprev = (s == 0) ? hz_bf : Yc0 + (size_t)(s - 1) * BATCH * CONH;
    c0.hp_stride = CONH;
    c0.whh = wC0H; c0.bias = bC0;
    c0.cprev = (s == 0) ? hz_f : cc0; c0.cnew = cc0;
    c0.hnew = Yc0 + (size_t)s * BATCH * CONH; c0.hn_stride = CONH;
    lstm_step<<<dim3(CONH / 128, 1), 256, 0, stream>>>(c0, c0, CONH, 0);
    DirArgs c1{};
    c1.x = Yc0 + (size_t)s * BATCH * CONH; c1.wih = wC1I;
    c1.hprev = (s == 0) ? hz_bf : Yc1 + (size_t)(s - 1) * BATCH * CONH;
    c1.hp_stride = CONH;
    c1.whh = wC1H; c1.bias = bC1;
    c1.cprev = (s == 0) ? hz_f : cc1; c1.cnew = cc1;
    c1.hnew = Yc1 + (size_t)s * BATCH * CONH; c1.hn_stride = CONH;
    lstm_step<<<dim3(CONH / 128, 1), 256, 0, stream>>>(c1, c1, CONH, CONH);
  }
  gemm(Yc1, CONH, wLin, CONH, F(31), feat, LATD, 4 * BATCH, LATD, CONH);
  conv(feat, feat_bf, 4 * BATCH * LATD);

  // ---- decoder: precompute teacher-input and z projections, then recurrence ----
  gemm(xprev_bf, DIN, wD0I, 1024, nullptr, xproj, 4096, T_SEQ * BATCH, 4096, DIN);
  gemm(feat_bf, DIN, wD0I + 512, 1024, nullptr, zproj, 4096, 4 * BATCH, 4096, DIN);
  for (int t = 0; t < T_SEQ; ++t) {
    int bar = t / 16;
    bool rs = (t % 16 == 0);
    if (rs)
      k_make_hz<<<lgrid(BATCH * DECH), 256, 0, stream>>>(
          feat + (size_t)bar * BATCH * LATD, hzd_bf, hzd_f, BATCH * DECH);
    DirArgs a0{};
    a0.ginA = xproj + (size_t)t * BATCH * 4096;
    a0.ginB = zproj + (size_t)bar * BATCH * 4096;
    a0.hprev = rs ? hzd_bf : Yd0 + (size_t)(t - 1) * BATCH * DECH;
    a0.hp_stride = DECH;
    a0.whh = wD0H; a0.bias = bD0;
    a0.cprev = rs ? hzd_f : cd0; a0.cnew = cd0;
    a0.hnew = Yd0 + (size_t)t * BATCH * DECH; a0.hn_stride = DECH;
    lstm_step<<<dim3(DECH / 128, 1), 256, 0, stream>>>(a0, a0, DECH, 0);
    DirArgs a1{};
    a1.x = Yd0 + (size_t)t * BATCH * DECH; a1.wih = wD1I;
    a1.hprev = rs ? hzd_bf : Yd1 + (size_t)(t - 1) * BATCH * DECH;
    a1.hp_stride = DECH;
    a1.whh = wD1H; a1.bias = bD1;
    a1.cprev = rs ? hzd_f : cd1; a1.cnew = cd1;
    a1.hnew = Yd1 + (size_t)t * BATCH * DECH; a1.hn_stride = DECH;
    lstm_step<<<dim3(DECH / 128, 1), 256, 0, stream>>>(a1, a1, DECH, DECH);
  }

  // ---- output head: logits GEMM over all steps, softmax + transpose ----
  gemm(Yd1, DECH, wOut, DECH, F(41), logits, DIN, T_SEQ * BATCH, DIN, DECH);
  k_softmax_out<<<dim3(T_SEQ * BATCH), 256, 0, stream>>>(logits, out_xprob);
}
